// MultiHeadedAttention_25881472926016
// MI455X (gfx1250) — matrix operand order repaired and checked by reading, not compiled, not run on hardware
//
#include <hip/hip_runtime.h>
#include <cstdint>

#define B_ 4
#define S_ 2048
#define E_ 1024
#define H_ 16
#define D_ 64
#define SCALE_ 0.125f
#define NEG_INF_ (-1000000000.0f)

typedef __attribute__((ext_vector_type(16))) __bf16 v16bf;
typedef __attribute__((ext_vector_type(2)))  __bf16 v2bf;
typedef __attribute__((ext_vector_type(2)))  float  v2f;
typedef __attribute__((ext_vector_type(8)))  float  v8f;
typedef unsigned int u32;

union FAcc    { v8f v;   float f[8]; };
union BF16x16 { v16bf v; u32 u[8]; __bf16 h[16]; };

// ---------------------------------------------------------------------------
// helpers
// ---------------------------------------------------------------------------
// Vector-typed conversion so ISel emits a single v_cvt_pk_bf16_f32 vdst,vA,vB.
__device__ __forceinline__ u32 packbf2(float a, float b) {
  union { v2bf h; u32 u; } x;
  v2f s; s.x = a; s.y = b;
  x.h = __builtin_convertvector(s, v2bf);
  return x.u;
}

// A-matrix 16x32 bf16 fragment: lanes 0-15 rows M=0..15 hold K=0..15,
// lanes 16-31 same rows, K=16..31.  Row-major source, stride lda (elems).
__device__ __forceinline__ v16bf load_a16x32(const __bf16* base, int lda, int lane) {
  BF16x16 f;
  const u32* row = (const u32*)(base + (lane & 15) * lda);
  const int ko = (lane & 16) ? 8 : 0;   // in u32 units (K offset 16 elems)
#pragma unroll
  for (int r = 0; r < 8; ++r) f.u[r] = row[ko + r];
  return f.v;
}

// B-matrix 32x16 bf16 fragment, source stored row-major as [n][k] (i.e. B^T),
// stride ldb: lane n = lane&15; lanes 0-15 hold K=0..15, lanes 16-31 K=16..31.
__device__ __forceinline__ v16bf load_bt16x32(const __bf16* base, int ldb, int lane) {
  BF16x16 f;
  const u32* row = (const u32*)(base + (lane & 15) * ldb);
  const int ko = (lane & 16) ? 8 : 0;   // in u32 units (K offset 16 elems)
#pragma unroll
  for (int r = 0; r < 8; ++r) f.u[r] = row[ko + r];
  return f.v;
}

// B-matrix 32x16 bf16 fragment, source stored row-major as [k][n], stride ldb.
__device__ __forceinline__ v16bf load_bk16x32(const __bf16* base, int ldb, int lane) {
  BF16x16 f;
  const int n  = lane & 15;
  const int kb = (lane & 16) ? 16 : 0;
#pragma unroll
  for (int e = 0; e < 16; ++e) f.h[e] = base[(kb + e) * ldb + n];
  return f.v;
}

__device__ __forceinline__ v8f wmma_bf16(v16bf a, v16bf b, v8f c) {
  return __builtin_amdgcn_wmma_f32_16x16x32_bf16(false, a, false, b, (short)0, c,
                                                 false, false);
}

// ---------------------------------------------------------------------------
// Kernel 0: pack int32 mask -> 1 bit per (b,q,k)
// ---------------------------------------------------------------------------
__global__ __launch_bounds__(256)
void mask_pack_kernel(const int* __restrict__ mask, u32* __restrict__ bits, int nwords) {
  int w = blockIdx.x * blockDim.x + threadIdx.x;
  if (w >= nwords) return;
  const int* p = mask + (size_t)w * 32;
  u32 v = 0;
#pragma unroll 8
  for (int i = 0; i < 32; ++i) v |= (p[i] != 0 ? 1u : 0u) << i;
  bits[w] = v;
}

// ---------------------------------------------------------------------------
// Kernel 1: fused QKV projection.  Y = (X @ W^T + b) * scale, bf16 output.
// Tiles: BM=128, BN=64, BK=32.  8 waves = 4(M) x 2(N), each wave 32x32 out.
// Double-buffered LDS; register staging; one barrier per K step.
// blockIdx.z selects projection (0=Q scaled, 1=K, 2=V).
// ---------------------------------------------------------------------------
__global__ __launch_bounds__(256)
void qkv_proj_kernel(const float* __restrict__ query, const float* __restrict__ key_,
                     const float* __restrict__ value,
                     const float* __restrict__ Wq, const float* __restrict__ bq,
                     const float* __restrict__ Wk, const float* __restrict__ bk,
                     const float* __restrict__ Wv, const float* __restrict__ bv,
                     __bf16* __restrict__ Qb, __bf16* __restrict__ Kb,
                     __bf16* __restrict__ Vb) {
  const int z = blockIdx.z;
  const float* X    = (z == 0) ? query : (z == 1) ? key_ : value;
  const float* W    = (z == 0) ? Wq : (z == 1) ? Wk : Wv;
  const float* bias = (z == 0) ? bq : (z == 1) ? bk : bv;
  __bf16*      Y    = (z == 0) ? Qb : (z == 1) ? Kb : Vb;
  const float scale = (z == 0) ? SCALE_ : 1.0f;

  __shared__ __bf16 As[2 * 128 * 40];  // ping-pong 128x32 A tiles, pad to 40
  __shared__ __bf16 Bs[2 * 64 * 40];   // ping-pong 64x32 B tiles (stored [n][k])

  const int tid = threadIdx.x, lane = tid & 31, wave = tid >> 5;
  const int waveM = wave >> 1, waveN = wave & 1;
  const int rowBase = blockIdx.x * 128, colBase = blockIdx.y * 64;
  const int arow = tid >> 1, ahalf = tid & 1;   // A: 128 rows x 2 half-rows
  const int brow = tid >> 2, bseg = tid & 3;    // B: 64 rows x 4 segments

  FAcc acc[2][2];
#pragma unroll
  for (int i = 0; i < 2; ++i)
#pragma unroll
    for (int j = 0; j < 2; ++j)
#pragma unroll
      for (int e = 0; e < 8; ++e) acc[i][j].f[e] = 0.f;

  float4 ra0, ra1, ra2, ra3, rb0, rb1;
  auto stage_load = [&](int kt) {
    const float4* asrc = (const float4*)(X + (size_t)(rowBase + arow) * E_ + kt + ahalf * 16);
    ra0 = asrc[0]; ra1 = asrc[1]; ra2 = asrc[2]; ra3 = asrc[3];
    const float4* bsrc = (const float4*)(W + (size_t)(colBase + brow) * E_ + kt + bseg * 8);
    rb0 = bsrc[0]; rb1 = bsrc[1];
  };
  auto stage_store = [&](int buf) {
    u32* ad = (u32*)(As + buf * (128 * 40) + arow * 40 + ahalf * 16);
    ad[0] = packbf2(ra0.x, ra0.y); ad[1] = packbf2(ra0.z, ra0.w);
    ad[2] = packbf2(ra1.x, ra1.y); ad[3] = packbf2(ra1.z, ra1.w);
    ad[4] = packbf2(ra2.x, ra2.y); ad[5] = packbf2(ra2.z, ra2.w);
    ad[6] = packbf2(ra3.x, ra3.y); ad[7] = packbf2(ra3.z, ra3.w);
    u32* bd = (u32*)(Bs + buf * (64 * 40) + brow * 40 + bseg * 8);
    bd[0] = packbf2(rb0.x, rb0.y); bd[1] = packbf2(rb0.z, rb0.w);
    bd[2] = packbf2(rb1.x, rb1.y); bd[3] = packbf2(rb1.z, rb1.w);
  };

  stage_load(0);
  for (int kt = 0; kt < E_; kt += 32) {
    const int buf = (kt >> 5) & 1;
    stage_store(buf);
    if (kt + 32 < E_) stage_load(kt + 32);  // overlap next-tile HBM with compute
    __syncthreads();

    const __bf16* ab = As + buf * (128 * 40);
    const __bf16* bb = Bs + buf * (64 * 40);
    v16bf a0 = load_a16x32(ab + (waveM * 32 +  0) * 40, 40, lane);
    v16bf a1 = load_a16x32(ab + (waveM * 32 + 16) * 40, 40, lane);
    v16bf b0 = load_bt16x32(bb + (waveN * 32 +  0) * 40, 40, lane);
    v16bf b1 = load_bt16x32(bb + (waveN * 32 + 16) * 40, 40, lane);
    acc[0][0].v = wmma_bf16(a0, b0, acc[0][0].v);
    acc[0][1].v = wmma_bf16(a0, b1, acc[0][1].v);
    acc[1][0].v = wmma_bf16(a1, b0, acc[1][0].v);
    acc[1][1].v = wmma_bf16(a1, b1, acc[1][1].v);
  }

#pragma unroll
  for (int mt = 0; mt < 2; ++mt)
#pragma unroll
    for (int nt = 0; nt < 2; ++nt) {
      const int c = colBase + waveN * 32 + nt * 16 + (lane & 15);
      const float bv_ = bias[c];
#pragma unroll
      for (int r = 0; r < 8; ++r) {
        const int row = rowBase + waveM * 32 + mt * 16 + ((lane & 16) ? r + 8 : r);
        Y[(size_t)row * E_ + c] = (__bf16)((acc[mt][nt].f[r] + bv_) * scale);
      }
    }
}

// ---------------------------------------------------------------------------
// Kernel 2: flash attention.  Grid (S/128, H, B); 8 waves, 16 q-rows per wave.
// K/V streamed in double-buffered 64-row LDS tiles; online softmax; P staged
// bf16 in a per-wave LDS slab (same-wave DS ordering needs no barrier).
// ---------------------------------------------------------------------------
__global__ __launch_bounds__(256)
void attention_kernel(const __bf16* __restrict__ Qb, const __bf16* __restrict__ Kb,
                      const __bf16* __restrict__ Vb, const u32* __restrict__ maskbits,
                      __bf16* __restrict__ X2) {
  const int bi = blockIdx.z, h = blockIdx.y;
  const int Qbase = blockIdx.x * 128;
  const int tid = threadIdx.x, lane = tid & 31, wave = tid >> 5;
  const int qw = Qbase + wave * 16;

  __shared__ __bf16 Ks[2 * 64 * 72];   // ping-pong [key][d], pad 64->72
  __shared__ __bf16 Vs[2 * 64 * 72];
  __shared__ __bf16 Ps[8 * 16 * 72];   // per-wave P slab [row][key]
  __bf16* Pw = Ps + wave * 16 * 72;

  // Q A-fragments (SCALE already folded in), kept in VGPRs for whole kernel.
  v16bf qf[2];
  {
    const __bf16* qrow = Qb + (size_t)(bi * S_ + qw + (lane & 15)) * E_ + h * 64;
    const u32* r32 = (const u32*)qrow;
    const int ko = (lane & 16) ? 8 : 0;
#pragma unroll
    for (int kc = 0; kc < 2; ++kc) {
      BF16x16 f;
#pragma unroll
      for (int r = 0; r < 8; ++r) {
        f.u[r]     = r32[kc * 16 + ko + r];
      }
      qf[kc] = f.v;
    }
  }

  float m_r[8], l_r[8];
  FAcc Oacc[4];
#pragma unroll
  for (int r = 0; r < 8; ++r) { m_r[r] = -1e30f; l_r[r] = 0.f; }
#pragma unroll
  for (int dt = 0; dt < 4; ++dt)
#pragma unroll
    for (int e = 0; e < 8; ++e) Oacc[dt].f[e] = 0.f;

  const int ldrow = tid >> 2, ldseg = tid & 3;  // cooperative 64x64 tile load

  uint4 rk0, rk1, rv0, rv1;
  auto stage_load = [&](int kb) {
    const size_t gro = (size_t)(bi * S_ + kb * 64 + ldrow) * E_ + h * 64 + ldseg * 16;
    const uint4* ks = (const uint4*)(Kb + gro);
    const uint4* vs = (const uint4*)(Vb + gro);
    rk0 = ks[0]; rk1 = ks[1];
    rv0 = vs[0]; rv1 = vs[1];
  };
  auto stage_store = [&](int buf) {
    uint4* kd = (uint4*)(Ks + buf * (64 * 72) + ldrow * 72 + ldseg * 16);
    uint4* vd = (uint4*)(Vs + buf * (64 * 72) + ldrow * 72 + ldseg * 16);
    kd[0] = rk0; kd[1] = rk1;
    vd[0] = rv0; vd[1] = rv1;
  };

  const int NB = S_ / 64;
  stage_load(0);
  for (int kb = 0; kb < NB; ++kb) {
    const int Kbase = kb * 64;
    const int buf = kb & 1;
    stage_store(buf);
    if (kb + 1 < NB) stage_load(kb + 1);  // overlap next K/V HBM with compute
    __syncthreads();

    const __bf16* kt = Ks + buf * (64 * 72);
    const __bf16* vt = Vs + buf * (64 * 72);

    // scores S = Q K^T  (16 x 64 per wave)
    FAcc sacc[4];
#pragma unroll
    for (int nt = 0; nt < 4; ++nt) {
#pragma unroll
      for (int e = 0; e < 8; ++e) sacc[nt].f[e] = 0.f;
#pragma unroll
      for (int kc = 0; kc < 2; ++kc) {
        v16bf kf = load_bt16x32(kt + nt * 16 * 72 + kc * 32, 72, lane);
        sacc[nt].v = wmma_bf16(qf[kc], kf, sacc[nt].v);
      }
    }

    // mask + online softmax (row stats replicated across the 16-lane group)
#pragma unroll
    for (int r = 0; r < 8; ++r) {
      const int row = (lane & 16) ? r + 8 : r;
      const int q = qw + row;
      const size_t wb = (((size_t)(bi * S_ + q) * S_) >> 5) + (Kbase >> 5);
      const u32 w0 = maskbits[wb], w1 = maskbits[wb + 1];
#pragma unroll
      for (int nt = 0; nt < 4; ++nt) {
        const u32 wv = (nt < 2) ? w0 : w1;
        if (!((wv >> (((nt & 1) << 4) + (lane & 15))) & 1u)) sacc[nt].f[r] = NEG_INF_;
      }
      float t = fmaxf(fmaxf(sacc[0].f[r], sacc[1].f[r]),
                      fmaxf(sacc[2].f[r], sacc[3].f[r]));
#pragma unroll
      for (int off = 8; off >= 1; off >>= 1) t = fmaxf(t, __shfl_xor(t, off, 32));
      const float mnew = fmaxf(m_r[r], t);
      const float corr = __expf(m_r[r] - mnew);
      m_r[r] = mnew;
      float rsum = 0.f;
#pragma unroll
      for (int nt = 0; nt < 4; ++nt) {
        const float p = __expf(sacc[nt].f[r] - mnew);
        rsum += p;
        Pw[row * 72 + nt * 16 + (lane & 15)] = (__bf16)p;
      }
#pragma unroll
      for (int off = 8; off >= 1; off >>= 1) rsum += __shfl_xor(rsum, off, 32);
      l_r[r] = l_r[r] * corr + rsum;
#pragma unroll
      for (int dt = 0; dt < 4; ++dt) Oacc[dt].f[r] *= corr;
    }

    // O += P @ V   (same-wave LDS writes->reads ordered via DScnt)
#pragma unroll
    for (int kc = 0; kc < 2; ++kc) {
      v16bf pa = load_a16x32(Pw + kc * 32, 72, lane);
#pragma unroll
      for (int dt = 0; dt < 4; ++dt) {
        v16bf vf = load_bk16x32(vt + kc * 32 * 72 + dt * 16, 72, lane);
        Oacc[dt].v = wmma_bf16(pa, vf, Oacc[dt].v);
      }
    }
    __syncthreads();
  }

  // normalize + store bf16 attention output in [B,S,E] layout
#pragma unroll
  for (int dt = 0; dt < 4; ++dt)
#pragma unroll
    for (int r = 0; r < 8; ++r) {
      const int row = (lane & 16) ? r + 8 : r;
      const float o = Oacc[dt].f[r] / l_r[r];
      X2[(size_t)(bi * S_ + qw + row) * E_ + h * 64 + dt * 16 + (lane & 15)] = (__bf16)o;
    }
}

// ---------------------------------------------------------------------------
// Kernel 3: output projection.  out = X2 @ Wo^T + bo, fp32 output.
// Same double-buffered structure as kernel 1; A is already bf16.
// ---------------------------------------------------------------------------
__global__ __launch_bounds__(256)
void out_proj_kernel(const __bf16* __restrict__ X2, const float* __restrict__ Wo,
                     const float* __restrict__ bo, float* __restrict__ out) {
  __shared__ __bf16 As[2 * 128 * 40];
  __shared__ __bf16 Bs[2 * 64 * 40];
  const int tid = threadIdx.x, lane = tid & 31, wave = tid >> 5;
  const int waveM = wave >> 1, waveN = wave & 1;
  const int rowBase = blockIdx.x * 128, colBase = blockIdx.y * 64;
  const int arow = tid >> 1, ahalf = tid & 1;
  const int brow = tid >> 2, bseg = tid & 3;

  FAcc acc[2][2];
#pragma unroll
  for (int i = 0; i < 2; ++i)
#pragma unroll
    for (int j = 0; j < 2; ++j)
#pragma unroll
      for (int e = 0; e < 8; ++e) acc[i][j].f[e] = 0.f;

  uint4 ra0, ra1;
  float4 rb0, rb1;
  auto stage_load = [&](int kt) {
    const uint4* asrc = (const uint4*)(X2 + (size_t)(rowBase + arow) * E_ + kt + ahalf * 16);
    ra0 = asrc[0]; ra1 = asrc[1];
    const float4* bsrc = (const float4*)(Wo + (size_t)(colBase + brow) * E_ + kt + bseg * 8);
    rb0 = bsrc[0]; rb1 = bsrc[1];
  };
  auto stage_store = [&](int buf) {
    uint4* ad = (uint4*)(As + buf * (128 * 40) + arow * 40 + ahalf * 16);
    ad[0] = ra0; ad[1] = ra1;
    u32* bd = (u32*)(Bs + buf * (64 * 40) + brow * 40 + bseg * 8);
    bd[0] = packbf2(rb0.x, rb0.y); bd[1] = packbf2(rb0.z, rb0.w);
    bd[2] = packbf2(rb1.x, rb1.y); bd[3] = packbf2(rb1.z, rb1.w);
  };

  stage_load(0);
  for (int kt = 0; kt < E_; kt += 32) {
    const int buf = (kt >> 5) & 1;
    stage_store(buf);
    if (kt + 32 < E_) stage_load(kt + 32);
    __syncthreads();

    const __bf16* ab = As + buf * (128 * 40);
    const __bf16* bb = Bs + buf * (64 * 40);
    v16bf a0 = load_a16x32(ab + (waveM * 32 +  0) * 40, 40, lane);
    v16bf a1 = load_a16x32(ab + (waveM * 32 + 16) * 40, 40, lane);
    v16bf b0 = load_bt16x32(bb + (waveN * 32 +  0) * 40, 40, lane);
    v16bf b1 = load_bt16x32(bb + (waveN * 32 + 16) * 40, 40, lane);
    acc[0][0].v = wmma_bf16(a0, b0, acc[0][0].v);
    acc[0][1].v = wmma_bf16(a0, b1, acc[0][1].v);
    acc[1][0].v = wmma_bf16(a1, b0, acc[1][0].v);
    acc[1][1].v = wmma_bf16(a1, b1, acc[1][1].v);
  }

#pragma unroll
  for (int mt = 0; mt < 2; ++mt)
#pragma unroll
    for (int nt = 0; nt < 2; ++nt) {
      const int c = colBase + waveN * 32 + nt * 16 + (lane & 15);
      const float bv_ = bo[c];
#pragma unroll
      for (int r = 0; r < 8; ++r) {
        const int row = rowBase + waveM * 32 + mt * 16 + ((lane & 16) ? r + 8 : r);
        out[(size_t)row * E_ + c] = acc[mt][nt].f[r] + bv_;
      }
    }
}

// ---------------------------------------------------------------------------
// launch
// ---------------------------------------------------------------------------
extern "C" void kernel_launch(void* const* d_in, const int* in_sizes, int n_in,
                              void* d_out, int out_size, void* d_ws, size_t ws_size,
                              hipStream_t stream) {
  (void)in_sizes; (void)n_in; (void)out_size; (void)ws_size;
  const float* query = (const float*)d_in[0];
  const float* key_  = (const float*)d_in[1];
  const float* value = (const float*)d_in[2];
  const int*   mask  = (const int*)d_in[3];
  const float* Wq = (const float*)d_in[4];
  const float* bq = (const float*)d_in[5];
  const float* Wk = (const float*)d_in[6];
  const float* bk = (const float*)d_in[7];
  const float* Wv = (const float*)d_in[8];
  const float* bv = (const float*)d_in[9];
  const float* Wo = (const float*)d_in[10];
  const float* bo = (const float*)d_in[11];
  float* out = (float*)d_out;

  // workspace layout: Qb|Kb|Vb|X2 (bf16, 16 MB each) + mask bitfield (2 MB)
  const size_t np = (size_t)B_ * S_ * E_;
  __bf16* Qb = (__bf16*)d_ws;
  __bf16* Kb = Qb + np;
  __bf16* Vb = Kb + np;
  __bf16* X2 = Vb + np;
  u32* maskbits = (u32*)(X2 + np);
  const int nwords = (int)((size_t)B_ * S_ * S_ / 32);

  mask_pack_kernel<<<dim3((nwords + 255) / 256), dim3(256), 0, stream>>>(mask, maskbits, nwords);
  qkv_proj_kernel<<<dim3((B_ * S_) / 128, E_ / 64, 3), dim3(256), 0, stream>>>(
      query, key_, value, Wq, bq, Wk, bk, Wv, bv, Qb, Kb, Vb);
  attention_kernel<<<dim3(S_ / 128, H_, B_), dim3(256), 0, stream>>>(
      Qb, Kb, Vb, maskbits, X2);
  out_proj_kernel<<<dim3((B_ * S_) / 128, E_ / 64), dim3(256), 0, stream>>>(
      X2, Wo, bo, out);
}
